// MultiHeadLatentAttention_42915313221712
// MI455X (gfx1250) — matrix operand order repaired and checked by reading, not compiled, not run on hardware
//
#include <hip/hip_runtime.h>
#include <hip/hip_bf16.h>
#include <math.h>

typedef __bf16 bf16_t;
typedef __attribute__((ext_vector_type(16))) __bf16 v16bf;
typedef __attribute__((ext_vector_type(8)))  __bf16 v8bf;
typedef __attribute__((ext_vector_type(8)))  float  v8f;
typedef __attribute__((ext_vector_type(4)))  int    v4i_t;

#define B_    2
#define S_    2048
#define DIM_  2048
#define H_    16
#define QL_   1536
#define KVL_  512
#define NOPE_ 128
#define ROPE_ 64
#define QKD_  192        /* NOPE + ROPE */
#define VD_   128
#define ROWS_ (B_ * S_)  /* 4096 */
#define SCALE_ 0.07216878364870323f   /* 192^-0.5 */
#define EPS_   1.1920928955078125e-07f

// ---- CDNA5 async global->LDS path (ASYNCcnt), with portable fallback -------
#if defined(__gfx1250__) && \
    __has_builtin(__builtin_amdgcn_global_load_async_to_lds_b128) && \
    __has_builtin(__builtin_amdgcn_s_wait_asynccnt)
#define HAVE_ASYNC_LDS 1
#else
#define HAVE_ASYNC_LDS 0
#endif

typedef __attribute__((address_space(1))) v4i_t* gptr_v4i;
typedef __attribute__((address_space(3))) v4i_t* lptr_v4i;

__device__ __forceinline__ void async_copy16(const bf16_t* __restrict__ g, bf16_t* l) {
#if HAVE_ASYNC_LDS
  __builtin_amdgcn_global_load_async_to_lds_b128(
      (gptr_v4i)g, (lptr_v4i)l, /*offset=*/0, /*cpol=*/0);
#else
  *(v8bf*)l = *(const v8bf*)g;
#endif
}

template <int N>
__device__ __forceinline__ void async_wait() {
#if HAVE_ASYNC_LDS
  __builtin_amdgcn_s_wait_asynccnt(N);
#endif
}

__device__ __forceinline__ v8f v8f_zero() {
  v8f z;
#pragma unroll
  for (int i = 0; i < 8; ++i) z[i] = 0.0f;
  return z;
}

// Build a 16-element bf16 fragment from two 8-element (16B) groups.
// CDNA5 16-bit A-layout: lane holds K in [base,base+8) and [base+16,base+24);
// B-layout: same K split as A (p1 = p0 + 16).
__device__ __forceinline__ v16bf mk16(const bf16_t* p0, const bf16_t* p1) {
  union { v16bf v; v8bf h[2]; } u;
  u.h[0] = *(const v8bf*)p0;
  u.h[1] = *(const v8bf*)p1;
  return u.v;
}

__device__ __forceinline__ v8f wmma_bf16(v16bf a, v16bf b, v8f c) {
  // D(f32 16x16) = A(bf16 16x32) * B(bf16 32x16) + C
  return __builtin_amdgcn_wmma_f32_16x16x32_bf16(
      /*neg_a=*/false, a, /*neg_b=*/false, b,
      /*c_mod=*/(short)0, c, /*reuse_a=*/false, /*reuse_b=*/false);
}

// ---------------------------------------------------------------- fp32 -> bf16
__global__ void k_cvt_bf16(const float* __restrict__ in, bf16_t* __restrict__ out, int n) {
  int i = (blockIdx.x * blockDim.x + threadIdx.x) * 4;
  if (i + 3 < n) {
    float4 f = *(const float4*)(in + i);
    out[i + 0] = (bf16_t)f.x; out[i + 1] = (bf16_t)f.y;
    out[i + 2] = (bf16_t)f.z; out[i + 3] = (bf16_t)f.w;
  } else {
    for (int j = i; j < n; ++j) out[j] = (bf16_t)in[j];
  }
}

// ------------------------------------------------- C(f32) = A(bf16) @ W(bf16)^T
// A: MxK row-major, W: NxK row-major (W rows are B-matrix columns).
// Block tile 128x128, K-step 32, double-buffered async LDS staging.
// 8 waves, each 32x64 (2x4 WMMA accumulators).
__global__ __launch_bounds__(256) void k_gemm_bf16nt(
    const bf16_t* __restrict__ A, const bf16_t* __restrict__ W,
    const float* __restrict__ bias, float* __restrict__ C,
    int M, int N, int K)
{
  __shared__ bf16_t lA[2][128 * 40];   // padded stride 40 halfs, 16B-aligned rows
  __shared__ bf16_t lB[2][128 * 40];
  const int tid  = threadIdx.x;
  const int lane = tid & 31, wid = tid >> 5;
  const int half = lane >> 4, l15 = lane & 15;
  const int m0 = blockIdx.y * 128, n0 = blockIdx.x * 128;
  const int wm = (wid >> 1) * 32, wn = (wid & 1) * 64;

  // Stage one 128x32 A tile + 128x32 B tile: 4 async ops per thread.
  auto stage = [&](int t, int buf) {
    const int k0 = t * 32;
#pragma unroll
    for (int i = 0; i < 2; ++i) {               // 512 chunks of 8 halfs each
      int c = tid + i * 256;
      int r = c >> 2, co = (c & 3) * 8;
      async_copy16(&A[(size_t)(m0 + r) * K + (k0 + co)], &lA[buf][r * 40 + co]);
      // Clamp (not skip) OOB rows so every wave issues the same ASYNCcnt ops;
      // clamped rows only feed output columns >= N, which are never stored.
      int rb = (n0 + r < N) ? (n0 + r) : (N - 1);
      async_copy16(&W[(size_t)rb * K + (k0 + co)], &lB[buf][r * 40 + co]);
    }
  };

  v8f acc[2][4];
#pragma unroll
  for (int i = 0; i < 2; ++i)
#pragma unroll
    for (int j = 0; j < 4; ++j) acc[i][j] = v8f_zero();

  const int T = K >> 5;                          // K multiple of 32
  stage(0, 0);
  for (int t = 0; t < T; ++t) {
    const int buf = t & 1;
    if (t + 1 < T) {
      stage(t + 1, (t + 1) & 1);                 // prefetch next tile
      async_wait<4>();                           // tile t resident (in-order)
    } else {
      async_wait<0>();
    }
    __syncthreads();                             // tile t visible to all waves

    v16bf af[2], bfr[4];
#pragma unroll
    for (int sm = 0; sm < 2; ++sm) {
      const bf16_t* p = &lA[buf][(wm + sm * 16 + l15) * 40 + half * 8];
      af[sm] = mk16(p, p + 16);
    }
#pragma unroll
    for (int sn = 0; sn < 4; ++sn) {
      const bf16_t* p = &lB[buf][(wn + sn * 16 + l15) * 40 + half * 8];
      bfr[sn] = mk16(p, p + 16);
    }
#pragma unroll
    for (int sm = 0; sm < 2; ++sm)
#pragma unroll
      for (int sn = 0; sn < 4; ++sn)
        acc[sm][sn] = wmma_bf16(af[sm], bfr[sn], acc[sm][sn]);

    __syncthreads();                             // all waves done reading buf
  }

#pragma unroll
  for (int sm = 0; sm < 2; ++sm)
#pragma unroll
    for (int sn = 0; sn < 4; ++sn)
#pragma unroll
      for (int v = 0; v < 8; ++v) {
        int r    = m0 + wm + sm * 16 + v + half * 8;   // C layout: M=v+8*half
        int cidx = n0 + wn + sn * 16 + l15;            //           N=lane&15
        if (cidx < N) {
          float val = acc[sm][sn][v];
          if (bias) val += bias[cidx];
          C[(size_t)r * N + cidx] = val;
        }
      }
}

// ----------------------------------------------------------------- RMSNorm row
__global__ __launch_bounds__(256) void k_rmsnorm(
    const float* __restrict__ in, int in_stride,
    const float* __restrict__ w,
    bf16_t* __restrict__ out, int out_stride, int D)
{
  __shared__ float red[256];
  const int row = blockIdx.x;
  const float* x = in + (size_t)row * in_stride;
  float ss = 0.0f;
  for (int c = threadIdx.x; c < D; c += 256) { float v = x[c]; ss += v * v; }
  red[threadIdx.x] = ss;
  __syncthreads();
  for (int s = 128; s > 0; s >>= 1) {
    if (threadIdx.x < s) red[threadIdx.x] += red[threadIdx.x + s];
    __syncthreads();
  }
  float inv = rsqrtf(red[0] / (float)D + EPS_);
  for (int c = threadIdx.x; c < D; c += 256)
    out[(size_t)row * out_stride + c] = (bf16_t)(x[c] * inv * w[c]);
}

// ------------------------------------------------------- RoPE on q (per head)
__global__ void k_rope_q(const float* __restrict__ q,
                         const float* __restrict__ fc, const float* __restrict__ fs,
                         bf16_t* __restrict__ out)
{
  const int row = blockIdx.x;
  const int s = row & (S_ - 1);
  const float* qr = q + (size_t)row * (H_ * QKD_);
  bf16_t* orow = out + (size_t)row * (H_ * QKD_);
  for (int e = threadIdx.x; e < H_ * QKD_; e += blockDim.x) {
    int h = e / QKD_, d = e - h * QKD_;
    float val;
    if (d < NOPE_) {
      val = qr[e];
    } else {
      int i = (d - NOPE_) >> 1;
      float c = fc[s * (ROPE_ / 2) + i], sn = fs[s * (ROPE_ / 2) + i];
      float x0 = qr[h * QKD_ + NOPE_ + 2 * i];
      float x1 = qr[h * QKD_ + NOPE_ + 2 * i + 1];
      val = ((d & 1) == 0) ? (x0 * c - x1 * sn) : (x0 * sn + x1 * c);
    }
    orow[e] = (bf16_t)val;
  }
}

// ----------------------------------------------------------- RoPE on k_pe (64)
__global__ void k_rope_kpe(const float* __restrict__ kv,
                           const float* __restrict__ fc, const float* __restrict__ fs,
                           bf16_t* __restrict__ kpe)
{
  const int row = blockIdx.x;
  const int s = row & (S_ - 1);
  int d = threadIdx.x;                      // 0..63
  int i = d >> 1;
  float c = fc[s * (ROPE_ / 2) + i], sn = fs[s * (ROPE_ / 2) + i];
  const float* kr = kv + (size_t)row * (KVL_ + ROPE_) + KVL_;
  float x0 = kr[2 * i], x1 = kr[2 * i + 1];
  float val = ((d & 1) == 0) ? (x0 * c - x1 * sn) : (x0 * sn + x1 * c);
  kpe[(size_t)row * ROPE_ + d] = (bf16_t)val;
}

// --------------------------- split kvb into K (nope|pe broadcast) and V, bf16
__global__ void k_split_kv(const float* __restrict__ kvb,
                           const bf16_t* __restrict__ kpe,
                           bf16_t* __restrict__ Kc, bf16_t* __restrict__ Vc)
{
  const int row = blockIdx.x;
  const float* r = kvb + (size_t)row * (H_ * (NOPE_ + VD_));
  bf16_t* ko = Kc + (size_t)row * (H_ * QKD_);
  bf16_t* vo = Vc + (size_t)row * (H_ * VD_);
  const bf16_t* kp = kpe + (size_t)row * ROPE_;
  for (int e = threadIdx.x; e < H_ * QKD_; e += blockDim.x) {
    int h = e / QKD_, d = e - h * QKD_;
    ko[e] = (d < NOPE_) ? (bf16_t)r[h * (NOPE_ + VD_) + d] : kp[d - NOPE_];
  }
  for (int e = threadIdx.x; e < H_ * VD_; e += blockDim.x) {
    int h = e / VD_, d = e - h * VD_;
    vo[e] = (bf16_t)r[h * (NOPE_ + VD_) + NOPE_ + d];
  }
}

// ----------------------------------------------------- flash attention (causal)
// grid: (S/128, H, B); 256 threads = 8 waves; wave w owns queries [blk*128+w*16, +16).
// Block-shared staging: K tile 32x192 (async DMA), V tile transposed 128x32,
// per-wave P tile 16x32 for the P(A-fragment) reshape.
__global__ __launch_bounds__(256) void k_flash_attn(
    const bf16_t* __restrict__ Q, const bf16_t* __restrict__ Kt,
    const bf16_t* __restrict__ V, bf16_t* __restrict__ O)
{
  __shared__ bf16_t ldsK[32 * 208];        // stride 208 halfs
  __shared__ bf16_t ldsVt[128 * 40];       // [d][key], stride 40 halfs
  __shared__ bf16_t ldsP[8][16 * 32];      // per-wave P tile

  const int tid  = threadIdx.x;
  const int lane = tid & 31, wid = tid >> 5;
  const int half = lane >> 4, l15 = lane & 15;
  const int qblk = blockIdx.x, h = blockIdx.y, b = blockIdx.z;
  const int s0 = qblk * 128 + wid * 16;
  const int s_max = s0 + 15;

  // Q fragments: 16 rows x 192 (6 K-chunks of 32)
  v16bf qf[6];
#pragma unroll
  for (int kc = 0; kc < 6; ++kc) {
    const bf16_t* p = Q + (size_t)(b * S_ + s0 + l15) * (H_ * QKD_)
                        + h * QKD_ + kc * 32 + half * 8;
    qf[kc] = mk16(p, p + 16);
  }

  v8f acco[8];
#pragma unroll
  for (int j = 0; j < 8; ++j) acco[j] = v8f_zero();
  float m_v[8], l_v[8];
#pragma unroll
  for (int v = 0; v < 8; ++v) { m_v[v] = -INFINITY; l_v[v] = 0.0f; }

  const int nkb = (qblk + 1) * 4;          // key blocks of 32 up to causal limit
  for (int kb = 0; kb < nkb; ++kb) {
    const int k0 = kb * 32;
    __syncthreads();                       // prior compute done reading LDS
    // stage K tile (32 keys x 192) via async DMA: 768 chunks of 8 halfs
#pragma unroll
    for (int i = 0; i < 3; ++i) {
      int c = tid + i * 256;
      int key = c / 24, off = (c % 24) * 8;
      async_copy16(&Kt[(size_t)(b * S_ + k0 + key) * (H_ * QKD_) + h * QKD_ + off],
                   &ldsK[key * 208 + off]);
    }
    // stage V tile transposed (d-major): 512 chunks of 8 halfs
#pragma unroll
    for (int i = 0; i < 2; ++i) {
      int c = tid + i * 256;
      int key = c >> 4, off = (c & 15) * 8;
      v8bf vv = *(const v8bf*)&V[(size_t)(b * S_ + k0 + key) * (H_ * VD_) + h * VD_ + off];
#pragma unroll
      for (int j = 0; j < 8; ++j) ldsVt[(off + j) * 40 + key] = vv[j];
    }
    async_wait<0>();
    __syncthreads();

    if (k0 <= s_max) {
      // scores: two 16x16 tiles (keys k0..+15, k0+16..+31)
      v8f sc[2];
#pragma unroll
      for (int c = 0; c < 2; ++c) {
        sc[c] = v8f_zero();
#pragma unroll
        for (int kc = 0; kc < 6; ++kc) {
          const bf16_t* p = &ldsK[(c * 16 + l15) * 208 + kc * 32 + half * 8];
          v16bf bfrag = mk16(p, p + 16);
          sc[c] = wmma_bf16(qf[kc], bfrag, sc[c]);
        }
      }
      // scale + causal mask (C layout: M=v+8*half, N=lane&15)
#pragma unroll
      for (int c = 0; c < 2; ++c)
#pragma unroll
        for (int v = 0; v < 8; ++v) {
          int key  = k0 + c * 16 + l15;
          int srow = s0 + v + half * 8;
          sc[c][v] = (key <= srow) ? sc[c][v] * SCALE_ : -INFINITY;
        }
      // row max (reduce across 16-lane half-groups)
      float mnew[8];
#pragma unroll
      for (int v = 0; v < 8; ++v) {
        float t = fmaxf(sc[0][v], sc[1][v]);
        t = fmaxf(t, __shfl_xor(t, 1, 32));
        t = fmaxf(t, __shfl_xor(t, 2, 32));
        t = fmaxf(t, __shfl_xor(t, 4, 32));
        t = fmaxf(t, __shfl_xor(t, 8, 32));
        mnew[v] = fmaxf(m_v[v], t);
      }
      // p = exp(s - m_new); row sums
#pragma unroll
      for (int c = 0; c < 2; ++c)
#pragma unroll
        for (int v = 0; v < 8; ++v)
          sc[c][v] = __expf(sc[c][v] - mnew[v]);
      float ladd[8];
#pragma unroll
      for (int v = 0; v < 8; ++v) {
        float t = sc[0][v] + sc[1][v];
        t += __shfl_xor(t, 1, 32);
        t += __shfl_xor(t, 2, 32);
        t += __shfl_xor(t, 4, 32);
        t += __shfl_xor(t, 8, 32);
        ladd[v] = t;
      }
      // rescale running state + accumulators
#pragma unroll
      for (int v = 0; v < 8; ++v) {
        float sold = __expf(m_v[v] - mnew[v]);
        l_v[v] = l_v[v] * sold + ladd[v];
        m_v[v] = mnew[v];
#pragma unroll
        for (int j = 0; j < 8; ++j) acco[j][v] *= sold;
      }
      // P -> LDS (bf16), reshape to A-fragment (wave-private, DS in-order)
#pragma unroll
      for (int c = 0; c < 2; ++c)
#pragma unroll
        for (int v = 0; v < 8; ++v) {
          int r = v + half * 8;
          ldsP[wid][r * 32 + c * 16 + l15] = (bf16_t)sc[c][v];
        }
      const bf16_t* pp = &ldsP[wid][l15 * 32 + half * 8];
      v16bf pf = mk16(pp, pp + 16);
      // O += P (16x32) x V (32x16 per d-chunk)
#pragma unroll
      for (int j = 0; j < 8; ++j) {
        const bf16_t* vp = &ldsVt[(j * 16 + l15) * 40 + half * 8];
        v16bf vfrag = mk16(vp, vp + 16);
        acco[j] = wmma_bf16(pf, vfrag, acco[j]);
      }
    }
  }

  // normalize and store bf16
#pragma unroll
  for (int j = 0; j < 8; ++j)
#pragma unroll
    for (int v = 0; v < 8; ++v) {
      int r = v + half * 8;
      size_t o = (size_t)(b * S_ + s0 + r) * (H_ * VD_) + h * VD_ + j * 16 + l15;
      O[o] = (bf16_t)(acco[j][v] / l_v[v]);
    }
}

// ------------------------------------------------------------------- launcher
extern "C" void kernel_launch(void* const* d_in, const int* in_sizes, int n_in,
                              void* d_out, int out_size, void* d_ws, size_t ws_size,
                              hipStream_t stream)
{
  const float* x     = (const float*)d_in[0];
  const float* fc    = (const float*)d_in[2];
  const float* fs    = (const float*)d_in[3];
  const float* wqa   = (const float*)d_in[5];
  const float* wqab  = (const float*)d_in[6];
  const float* qnw   = (const float*)d_in[7];
  const float* wqb   = (const float*)d_in[8];
  const float* wqbb  = (const float*)d_in[9];
  const float* wkva  = (const float*)d_in[10];
  const float* wkvab = (const float*)d_in[11];
  const float* kvnw  = (const float*)d_in[12];
  const float* wkvb  = (const float*)d_in[13];
  const float* wkvbb = (const float*)d_in[14];
  const float* wo    = (const float*)d_in[15];
  const float* wob   = (const float*)d_in[16];
  float* out = (float*)d_out;

  char* ws = (char*)d_ws;
  size_t off = 0;
  auto alloc = [&](size_t bytes) -> void* {
    void* p = ws + off;
    off += (bytes + 255) & ~(size_t)255;
    return p;
  };
  bf16_t* xb     = (bf16_t*)alloc((size_t)ROWS_ * DIM_ * 2);
  bf16_t* wqa_h  = (bf16_t*)alloc((size_t)QL_ * DIM_ * 2);
  bf16_t* wqb_h  = (bf16_t*)alloc((size_t)H_ * QKD_ * QL_ * 2);
  bf16_t* wkva_h = (bf16_t*)alloc((size_t)(KVL_ + ROPE_) * DIM_ * 2);
  bf16_t* wkvb_h = (bf16_t*)alloc((size_t)H_ * (NOPE_ + VD_) * KVL_ * 2);
  bf16_t* wo_h   = (bf16_t*)alloc((size_t)DIM_ * H_ * VD_ * 2);
  float*  qa     = (float*)alloc((size_t)ROWS_ * QL_ * 4);
  bf16_t* qn     = (bf16_t*)alloc((size_t)ROWS_ * QL_ * 2);
  float*  qf32   = (float*)alloc((size_t)ROWS_ * H_ * QKD_ * 4);
  bf16_t* qb     = (bf16_t*)alloc((size_t)ROWS_ * H_ * QKD_ * 2);
  float*  kva    = (float*)alloc((size_t)ROWS_ * (KVL_ + ROPE_) * 4);
  bf16_t* kvn    = (bf16_t*)alloc((size_t)ROWS_ * KVL_ * 2);
  bf16_t* kpe    = (bf16_t*)alloc((size_t)ROWS_ * ROPE_ * 2);
  float*  kvbf   = (float*)alloc((size_t)ROWS_ * H_ * (NOPE_ + VD_) * 4);
  bf16_t* kb     = (bf16_t*)alloc((size_t)ROWS_ * H_ * QKD_ * 2);
  bf16_t* vb     = (bf16_t*)alloc((size_t)ROWS_ * H_ * VD_ * 2);
  bf16_t* ao     = (bf16_t*)alloc((size_t)ROWS_ * H_ * VD_ * 2);

  auto cvt = [&](const float* src, bf16_t* dst, int n) {
    int blocks = (n / 4 + 255) / 256;
    k_cvt_bf16<<<blocks, 256, 0, stream>>>(src, dst, n);
  };
  cvt(x, xb, ROWS_ * DIM_);
  cvt(wqa, wqa_h, QL_ * DIM_);
  cvt(wqb, wqb_h, H_ * QKD_ * QL_);
  cvt(wkva, wkva_h, (KVL_ + ROPE_) * DIM_);
  cvt(wkvb, wkvb_h, H_ * (NOPE_ + VD_) * KVL_);
  cvt(wo, wo_h, DIM_ * H_ * VD_);

  auto gemm = [&](const bf16_t* A, const bf16_t* W, const float* bias,
                  float* C, int M, int N, int K) {
    dim3 g((N + 127) / 128, M / 128);
    k_gemm_bf16nt<<<g, 256, 0, stream>>>(A, W, bias, C, M, N, K);
  };

  // q_a = x @ wq_a^T + b ; rmsnorm -> bf16
  gemm(xb, wqa_h, wqab, qa, ROWS_, QL_, DIM_);
  k_rmsnorm<<<ROWS_, 256, 0, stream>>>(qa, QL_, qnw, qn, QL_, QL_);
  // q = qn @ wq_b^T + b ; rope -> bf16 (b,s,h,192)
  gemm(qn, wqb_h, wqbb, qf32, ROWS_, H_ * QKD_, QL_);
  k_rope_q<<<ROWS_, 256, 0, stream>>>(qf32, fc, fs, qb);
  // kv = x @ wkv_a^T + b ; rmsnorm(kv_c) -> bf16 ; rope(k_pe) -> bf16
  gemm(xb, wkva_h, wkvab, kva, ROWS_, KVL_ + ROPE_, DIM_);
  k_rmsnorm<<<ROWS_, 256, 0, stream>>>(kva, KVL_ + ROPE_, kvnw, kvn, KVL_, KVL_);
  k_rope_kpe<<<ROWS_, ROPE_, 0, stream>>>(kva, fc, fs, kpe);
  // kvb = kvn @ wkv_b^T + b ; split into K (nope|pe) and V
  gemm(kvn, wkvb_h, wkvbb, kvbf, ROWS_, H_ * (NOPE_ + VD_), KVL_);
  k_split_kv<<<ROWS_, 256, 0, stream>>>(kvbf, kpe, kb, vb);
  // causal flash attention
  dim3 ag(S_ / 128, H_, B_);
  k_flash_attn<<<ag, 256, 0, stream>>>(qb, kb, vb, ao);
  // out = attn @ wo^T + b (fp32 into d_out)
  gemm(ao, wo_h, wob, out, ROWS_, DIM_, H_ * VD_);

  (void)in_sizes; (void)n_in; (void)out_size; (void)ws_size;
}
